// CausalSelfAttention_38053410242622
// MI455X (gfx1250) — matrix operand order repaired and checked by reading, not compiled, not run on hardware
//
#include <hip/hip_runtime.h>
#include <hip/hip_bf16.h>

typedef __attribute__((ext_vector_type(16))) _Float16 v16h;
typedef __attribute__((ext_vector_type(8)))  _Float16 v8h;
typedef __attribute__((ext_vector_type(8)))  float    v8f;
typedef __attribute__((ext_vector_type(4)))  int      v4i;

#define N_B   2
#define N_T   2048
#define N_C   1024
#define N_H   16
#define N_D   64
#define N_BT  (N_B * N_T)      // 4096
#define N_3C  (3 * N_C)        // 3072

#define LDS_PITCH 40           // 32 data halfs + 8 pad halfs (16B-aligned rows)

#if defined(__has_builtin)
#if __has_builtin(__builtin_amdgcn_global_load_async_to_lds_b128) && \
    __has_builtin(__builtin_amdgcn_s_wait_asynccnt)
#define USE_ASYNC_LDS 1
#endif
#endif

// ---------------- fragment helpers ----------------
__device__ __forceinline__ v16h mk16(v8h lo, v8h hi) {
  v16h r;
#pragma unroll
  for (int i = 0; i < 8; ++i) { r[i] = lo[i]; r[i + 8] = hi[i]; }
  return r;
}

// A fragment (16x32 f16, MxK): lane m = lane&15, g = lane>>4,
// element e -> K = 16*g + e  => two contiguous 16B loads.
__device__ __forceinline__ v16h load_a16(const _Float16* rowptr, int g) {
  v8h lo = *(const v8h*)(rowptr + 16 * g);
  v8h hi = *(const v8h*)(rowptr + 16 * g + 8);
  return mk16(lo, hi);
}

// B fragment (32x16 f16, KxN): lane n = lane&15, g = lane>>4,
// element e -> K = 16*g + e  => one contiguous 32B region per lane.
__device__ __forceinline__ v16h load_b16(const _Float16* colptr, int g) {
  v8h lo = *(const v8h*)(colptr + 16 * g);
  v8h hi = *(const v8h*)(colptr + 16 * g + 8);
  return mk16(lo, hi);
}

__device__ __forceinline__ v8f wmma_f16(v16h a, v16h b, v8f c) {
  return __builtin_amdgcn_wmma_f32_16x16x32_f16(false, a, false, b, (short)0, c,
                                                false, false);
}

// ---------------- async LDS staging ----------------
__device__ __forceinline__ void async_copy16(const _Float16* g, _Float16* l) {
#if USE_ASYNC_LDS
  __builtin_amdgcn_global_load_async_to_lds_b128(
      (__attribute__((address_space(1))) v4i*)g,
      (__attribute__((address_space(3))) v4i*)l, 0, 0);
#else
  *(v8h*)l = *(const v8h*)g;
#endif
}

template <int Ncnt>
__device__ __forceinline__ void wait_async() {
#if USE_ASYNC_LDS
  __builtin_amdgcn_s_wait_asynccnt((unsigned short)Ncnt);
#endif
}

// ---------------- scale / quantize ----------------
__global__ void zero_sums_kernel(float* sums) {
  if (threadIdx.x == 0) { sums[0] = 0.f; sums[1] = 0.f; }
}

__global__ void abs_sum_kernel(const float* __restrict__ w, int nelem,
                               float* __restrict__ out) {
  __shared__ float red[256];
  float s = 0.f;
  for (int i = blockIdx.x * blockDim.x + threadIdx.x; i < nelem;
       i += gridDim.x * blockDim.x)
    s += fabsf(w[i]);
  red[threadIdx.x] = s;
  __syncthreads();
  for (int off = 128; off > 0; off >>= 1) {
    if ((int)threadIdx.x < off) red[threadIdx.x] += red[threadIdx.x + off];
    __syncthreads();
  }
  if (threadIdx.x == 0) atomicAdd(out, red[0]);
}

__global__ void quantize_w_kernel(const float* __restrict__ w,
                                  _Float16* __restrict__ wq,
                                  const float* __restrict__ sum, float inv_n,
                                  int nelem) {
  float scale = fmaxf(sum[0] * inv_n, 1e-8f);
  float inv_scale = 1.f / scale;
  for (int i = blockIdx.x * blockDim.x + threadIdx.x; i < nelem;
       i += gridDim.x * blockDim.x) {
    float q = rintf(w[i] * inv_scale);  // round-half-even like jnp.round
    q = fminf(fmaxf(q, -2.f), 2.f);
    wq[i] = (_Float16)(q * scale);
  }
}

__global__ void cvt_f32_f16_kernel(const float* __restrict__ x,
                                   _Float16* __restrict__ y, int nelem) {
  for (int i = blockIdx.x * blockDim.x + threadIdx.x; i < nelem;
       i += gridDim.x * blockDim.x)
    y[i] = (_Float16)x[i];
}

// ---------------- QKV GEMM: [4096,1024] x [3072,1024]^T ----------------
// Block tile 128M x 128N, 8 waves (16 rows each), weight tile double-buffered
// in LDS via async global->LDS DMA. Epilogue scatters q/k/v to [B,H,T,D] f16.
__global__ void __launch_bounds__(256)
gemm_qkv_kernel(const _Float16* __restrict__ X, const _Float16* __restrict__ W,
                _Float16* __restrict__ Q, _Float16* __restrict__ K,
                _Float16* __restrict__ V) {
  __shared__ __align__(16) _Float16 bt[2][128 * LDS_PITCH];
  const int tid = threadIdx.x;
  const int lane = tid & 31, w = tid >> 5;
  const int nblk = N_3C / 128;  // 24
  const int bm = blockIdx.x / nblk, bn = blockIdx.x % nblk;
  const int m0 = bm * 128 + w * 16;
  const int o0 = bn * 128;
  const int n = lane & 15, g = lane >> 4;

  // staging map: each thread owns one 32B chunk of a weight row
  const int fr = tid >> 1;              // 0..127: row within the 128-col tile
  const int fc = (tid & 1) * 16;        // 0 or 16 halfs within the 32-K slab
  const _Float16* gsrc = W + (size_t)(o0 + fr) * N_C + fc;
  _Float16* lbase0 = &bt[0][fr * LDS_PITCH + fc];
  _Float16* lbase1 = &bt[1][fr * LDS_PITCH + fc];

  // prologue: stage kb = 0 into buffer 0
  async_copy16(gsrc + 0, lbase0);
  async_copy16(gsrc + 8, lbase0 + 8);

  v8f acc[8] = {};
  const _Float16* aptr = X + (size_t)(m0 + n) * N_C;

  for (int kb = 0; kb < N_C; kb += 32) {
    const int cur = (kb >> 5) & 1;
    if (kb + 32 < N_C) {
      _Float16* ld2 = cur ? lbase0 : lbase1;
      const _Float16* gs2 = gsrc + kb + 32;
      async_copy16(gs2 + 0, ld2);
      async_copy16(gs2 + 8, ld2 + 8);
      wait_async<2>();   // in-order completion => current buffer is ready
    } else {
      wait_async<0>();
    }
    __syncthreads();
    __builtin_prefetch(aptr + kb + 128, 0, 0);
    v16h a = load_a16(aptr + kb, g);
    const _Float16* btile = &bt[cur][0];
#pragma unroll
    for (int t = 0; t < 8; ++t) {
      const _Float16* bp = btile + (t * 16 + n) * LDS_PITCH + 16 * g;
      v16h b = mk16(*(const v8h*)bp, *(const v8h*)(bp + 8));
      acc[t] = wmma_f16(a, b, acc[t]);
    }
    __syncthreads();  // protect buffer from next iteration's DMA
  }
#pragma unroll
  for (int t = 0; t < 8; ++t) {
    int o = o0 + t * 16 + n;
    int sel = o >> 10, oc = o & (N_C - 1), h = oc >> 6, d = oc & 63;
    _Float16* dst = (sel == 0) ? Q : ((sel == 1) ? K : V);
#pragma unroll
    for (int r = 0; r < 8; ++r) {
      int m = m0 + r + 8 * g;
      int bb = m >> 11, tt = m & (N_T - 1);
      dst[(((size_t)(bb * N_H + h)) * N_T + tt) * N_D + d] = (_Float16)acc[t][r];
    }
  }
}

// ---------------- flash attention ----------------
// grid = B*H * (T/128); 8 waves/block, each wave owns 16 query rows
__global__ void __launch_bounds__(256)
attn_kernel(const _Float16* __restrict__ Qm, const _Float16* __restrict__ Km,
            const _Float16* __restrict__ Vm, _Float16* __restrict__ Y) {
  __shared__ __align__(16) _Float16 plds[8][16][32];
  const int lane = threadIdx.x & 31;
  const int w = threadIdx.x >> 5;
  const int bh = blockIdx.x >> 4;  // 0..31
  const int qb = blockIdx.x & 15;  // 0..15
  const int q0 = qb * 128 + w * 16;
  const int n = lane & 15, g = lane >> 4;

  const _Float16* qbase = Qm + ((size_t)bh * N_T + q0 + n) * N_D;
  const v16h aq0 = load_a16(qbase, g);       // head-dim K = 0..31
  const v16h aq1 = load_a16(qbase + 32, g);  // head-dim K = 32..63

  v8f oacc[4] = {};
  float mi[8], li[8];
#pragma unroll
  for (int r = 0; r < 8; ++r) { mi[r] = -3.0e38f; li[r] = 0.f; }

  const int kend = q0 + 16;  // causal bound (exclusive)
  for (int kc = 0; kc < kend; kc += 32) {
    // S = Q K^T for 32 keys (two 16-wide column groups)
    v8f s[2] = {};
#pragma unroll
    for (int f = 0; f < 2; ++f) {
      const _Float16* kb = Km + ((size_t)bh * N_T + kc + f * 16 + n) * N_D;
      s[f] = wmma_f16(aq0, load_b16(kb, g), s[f]);
      s[f] = wmma_f16(aq1, load_b16(kb + 32, g), s[f]);
    }
    // scale + causal mask
#pragma unroll
    for (int f = 0; f < 2; ++f) {
      int key = kc + f * 16 + n;
#pragma unroll
      for (int r = 0; r < 8; ++r) {
        int qrow = q0 + r + 8 * g;
        float sv = s[f][r] * 0.125f;  // 1/sqrt(64)
        s[f][r] = (key <= qrow) ? sv : -3.0e38f;
      }
    }
    // online softmax, rows live across 16-lane half-groups
    float pv0[8], pv1[8];
#pragma unroll
    for (int r = 0; r < 8; ++r) {
      float mx = fmaxf(s[0][r], s[1][r]);
#pragma unroll
      for (int sh = 1; sh < 16; sh <<= 1) mx = fmaxf(mx, __shfl_xor(mx, sh, 32));
      float mnew = fmaxf(mi[r], mx);
      float p0 = __expf(s[0][r] - mnew);
      float p1 = __expf(s[1][r] - mnew);
      float ps = p0 + p1;
#pragma unroll
      for (int sh = 1; sh < 16; sh <<= 1) ps += __shfl_xor(ps, sh, 32);
      float alpha = __expf(mi[r] - mnew);
      li[r] = li[r] * alpha + ps;
      mi[r] = mnew;
#pragma unroll
      for (int t = 0; t < 4; ++t) oacc[t][r] *= alpha;
      pv0[r] = p0; pv1[r] = p1;
    }
    // transpose P (D-fragment -> A-fragment) through LDS
#pragma unroll
    for (int r = 0; r < 8; ++r) {
      plds[w][r + 8 * g][n]      = (_Float16)pv0[r];
      plds[w][r + 8 * g][16 + n] = (_Float16)pv1[r];
    }
    __builtin_amdgcn_fence(__ATOMIC_ACQ_REL, "wavefront");
    __builtin_amdgcn_wave_barrier();
    v16h ap = load_a16(&plds[w][n][0], g);
    // O += P V (contraction over the 32 keys of this chunk)
#pragma unroll
    for (int t = 0; t < 4; ++t) {
      const _Float16* vb =
          Vm + ((size_t)bh * N_T + kc + 16 * g) * N_D + t * 16 + n;
      v16h bv;
#pragma unroll
      for (int e = 0; e < 16; ++e) bv[e] = vb[(size_t)e * N_D];
      oacc[t] = wmma_f16(ap, bv, oacc[t]);
    }
  }
  // epilogue: y[b, t, h*64+d] f16
  const int b = bh >> 4, h = bh & 15;
#pragma unroll
  for (int t = 0; t < 4; ++t) {
#pragma unroll
    for (int r = 0; r < 8; ++r) {
      int qrow = q0 + r + 8 * g;
      float val = oacc[t][r] / li[r];
      Y[((size_t)(b * N_T + qrow)) * N_C + h * N_D + t * 16 + n] = (_Float16)val;
    }
  }
}

// ---------------- projection GEMM: [4096,1024] x [1024,1024]^T -> fp32 ----
__global__ void __launch_bounds__(256)
gemm_proj_kernel(const _Float16* __restrict__ Yh, const _Float16* __restrict__ W,
                 float* __restrict__ Out) {
  __shared__ __align__(16) _Float16 bt[2][128 * LDS_PITCH];
  const int tid = threadIdx.x;
  const int lane = tid & 31, w = tid >> 5;
  const int nblk = N_C / 128;  // 8
  const int bm = blockIdx.x / nblk, bn = blockIdx.x % nblk;
  const int m0 = bm * 128 + w * 16;
  const int o0 = bn * 128;
  const int n = lane & 15, g = lane >> 4;

  const int fr = tid >> 1;
  const int fc = (tid & 1) * 16;
  const _Float16* gsrc = W + (size_t)(o0 + fr) * N_C + fc;
  _Float16* lbase0 = &bt[0][fr * LDS_PITCH + fc];
  _Float16* lbase1 = &bt[1][fr * LDS_PITCH + fc];

  async_copy16(gsrc + 0, lbase0);
  async_copy16(gsrc + 8, lbase0 + 8);

  v8f acc[8] = {};
  const _Float16* aptr = Yh + (size_t)(m0 + n) * N_C;

  for (int kb = 0; kb < N_C; kb += 32) {
    const int cur = (kb >> 5) & 1;
    if (kb + 32 < N_C) {
      _Float16* ld2 = cur ? lbase0 : lbase1;
      const _Float16* gs2 = gsrc + kb + 32;
      async_copy16(gs2 + 0, ld2);
      async_copy16(gs2 + 8, ld2 + 8);
      wait_async<2>();
    } else {
      wait_async<0>();
    }
    __syncthreads();
    __builtin_prefetch(aptr + kb + 128, 0, 0);
    v16h a = load_a16(aptr + kb, g);
    const _Float16* btile = &bt[cur][0];
#pragma unroll
    for (int t = 0; t < 8; ++t) {
      const _Float16* bp = btile + (t * 16 + n) * LDS_PITCH + 16 * g;
      v16h b = mk16(*(const v8h*)bp, *(const v8h*)(bp + 8));
      acc[t] = wmma_f16(a, b, acc[t]);
    }
    __syncthreads();
  }
#pragma unroll
  for (int t = 0; t < 8; ++t) {
    int o = o0 + t * 16 + n;
#pragma unroll
    for (int r = 0; r < 8; ++r) {
      int m = m0 + r + 8 * g;
      Out[(size_t)m * N_C + o] = acc[t][r];
    }
  }
}

// ---------------- launch ----------------
extern "C" void kernel_launch(void* const* d_in, const int* in_sizes, int n_in,
                              void* d_out, int out_size, void* d_ws,
                              size_t ws_size, hipStream_t stream) {
  const float* x      = (const float*)d_in[0];  // [2,2048,1024]
  const float* w_attn = (const float*)d_in[1];  // [3072,1024]
  const float* w_proj = (const float*)d_in[2];  // [1024,1024]
  float* out = (float*)d_out;                   // [2,2048,1024]

  char* ws = (char*)d_ws;
  size_t off = 0;
  float* sums = (float*)(ws + off);          off += 256;
  _Float16* wq_attn = (_Float16*)(ws + off); off += (size_t)N_3C * N_C * 2;
  _Float16* wq_proj = (_Float16*)(ws + off); off += (size_t)N_C * N_C * 2;
  _Float16* x16     = (_Float16*)(ws + off); off += (size_t)N_BT * N_C * 2;
  _Float16* q16     = (_Float16*)(ws + off); off += (size_t)N_BT * N_C * 2;
  _Float16* k16     = (_Float16*)(ws + off); off += (size_t)N_BT * N_C * 2;
  _Float16* v16_    = (_Float16*)(ws + off); off += (size_t)N_BT * N_C * 2;
  _Float16* y16     = (_Float16*)(ws + off); off += (size_t)N_BT * N_C * 2;

  const int n_wa = N_3C * N_C;  // 3145728
  const int n_wp = N_C * N_C;   // 1048576
  const int n_x  = N_BT * N_C;  // 4194304

  zero_sums_kernel<<<1, 64, 0, stream>>>(sums);
  abs_sum_kernel<<<512, 256, 0, stream>>>(w_attn, n_wa, &sums[0]);
  abs_sum_kernel<<<512, 256, 0, stream>>>(w_proj, n_wp, &sums[1]);
  quantize_w_kernel<<<2048, 256, 0, stream>>>(w_attn, wq_attn, &sums[0],
                                              1.f / (float)n_wa, n_wa);
  quantize_w_kernel<<<2048, 256, 0, stream>>>(w_proj, wq_proj, &sums[1],
                                              1.f / (float)n_wp, n_wp);
  cvt_f32_f16_kernel<<<2048, 256, 0, stream>>>(x, x16, n_x);

  // QKV: (4096/128) x (3072/128) = 32*24 = 768 blocks
  gemm_qkv_kernel<<<768, 256, 0, stream>>>(x16, wq_attn, q16, k16, v16_);
  // attention: B*H*(T/128) = 512 blocks
  attn_kernel<<<512, 256, 0, stream>>>(q16, k16, v16_, y16);
  // projection: (4096/128) x (1024/128) = 32*8 = 256 blocks
  gemm_proj_kernel<<<256, 256, 0, stream>>>(y16, wq_proj, out);
}
